// MinGRUCell_77309411811
// MI455X (gfx1250) — matrix operand order repaired and checked by reading, not compiled, not run on hardware
//
#include <hip/hip_runtime.h>

// Problem constants
#define Bdim 8
#define Ldim 4096
#define Ddim 1024
#define NCHUNK 64   // chunks along L for the parallel scan
#define CLEN   64   // chunk length (NCHUNK*CLEN == Ldim)

typedef __attribute__((ext_vector_type(16))) __bf16 v16bf;
typedef __attribute__((ext_vector_type(8)))  float  v8f;

__device__ __forceinline__ unsigned short f2bf(float f) {
  unsigned u = __float_as_uint(f);
  return (unsigned short)((u + 0x7FFFu + ((u >> 16) & 1u)) >> 16);  // RNE
}
__device__ __forceinline__ float bf2f(unsigned short h) {
  return __uint_as_float(((unsigned)h) << 16);
}

// ---------------- fp32 -> bf16 conversion (vectorized x4) ----------------
__global__ __launch_bounds__(256) void cvt4_kernel(const float* __restrict__ src,
                                                   unsigned short* __restrict__ dst,
                                                   int n4) {
  int t = blockIdx.x * 256 + threadIdx.x;
  if (t >= n4) return;
  float4 v = ((const float4*)src)[t];
  uint2 o;
  o.x = (unsigned)f2bf(v.x) | ((unsigned)f2bf(v.y) << 16);
  o.y = (unsigned)f2bf(v.z) | ((unsigned)f2bf(v.w) << 16);
  ((uint2*)dst)[t] = o;
}

// ---------------- async global -> LDS (CDNA5 path, ASYNCcnt) ----------------
// GVS mode: LDS address from the vdst VGPR, global address = 64-bit SGPR base + 32-bit VGPR offset
__device__ __forceinline__ void async_b128(unsigned ldsaddr, unsigned goff,
                                           const unsigned short* base) {
  asm volatile("global_load_async_to_lds_b128 %0, %1, %2"
               :: "v"(ldsaddr), "v"(goff), "s"(base) : "memory");
}
// low 32 bits of a generic pointer to LDS == LDS byte address (flat aperture mapping)
__device__ __forceinline__ unsigned lds_addr32(const void* p) {
  return (unsigned)(unsigned long long)p;
}

// ---------------- WMMA fragment loads from LDS ----------------
// A (16x32 bf16): lanes 0-15 hold M=r: K = 0..7 then 16..23 ; lanes 16-31: K = 8..15 then 24..31
__device__ __forceinline__ v16bf load_afrag(const unsigned short* row, int half) {
  v16bf f;
  ((uint4*)&f)[0] = *(const uint4*)(row + half * 8);
  ((uint4*)&f)[1] = *(const uint4*)(row + 16 + half * 8);
  return f;
}
// B (32x16 bf16): lane n (0-15) holds col n, K=0..7 then 16..23 ; lane n+16 holds col n, K=8..15 then 24..31
// B(k,n) = W[n][k] -> W rows are K-contiguous, perfect fit.
__device__ __forceinline__ v16bf load_bfrag(const unsigned short* row, int half) {
  v16bf f;
  ((uint4*)&f)[0] = *(const uint4*)(row + half * 8);
  ((uint4*)&f)[1] = *(const uint4*)(row + 16 + half * 8);
  return f;
}

// ---------------- Fused dual GEMM: g = sigm(xWg^T+bg), c = tanh(xWh^T+bh) ----------------
// grid = (N/64, M/128), block = 256 (8 waves). Wave (wm,wn) does a 32x32 subtile, both paths.
// Double-buffered LDS, async-to-LDS pipelined staging, 2x software pipeline unroll so all
// buffer indices are compile-time constants (no cndmask / branch overhead in the hot loop).
__global__ __launch_bounds__(256) void gemm_gate_cand(
    const unsigned short* __restrict__ xb,    // (32768, 1024) bf16
    const unsigned short* __restrict__ wgb,   // (1024, 1024) bf16, row e contiguous in d
    const unsigned short* __restrict__ whb,
    const float* __restrict__ bg,
    const float* __restrict__ bh,
    unsigned short* __restrict__ aG,          // gate a = g   (bf16)
    float* __restrict__ bOut)                 // b = (1-g)*c  (fp32, into d_out)
{
  __shared__ alignas(16) unsigned short sX [2][128][40];
  __shared__ alignas(16) unsigned short sWg[2][ 64][40];
  __shared__ alignas(16) unsigned short sWh[2][ 64][40];

  const int tid  = threadIdx.x;
  const int lane = tid & 31;
  const int wave = tid >> 5;
  const int wm   = wave & 3;          // 0..3 -> M offset wm*32
  const int wn   = wave >> 2;         // 0..1 -> N offset wn*32
  const int r    = lane & 15;
  const int half = lane >> 4;
  const int m0   = blockIdx.y * 128;
  const int n0   = blockIdx.x * 64;

  v8f accg[2][2] = {};
  v8f acch[2][2] = {};

  const int ldrow  = tid >> 2;        // 0..63
  const int ldchnk = (tid & 3) * 8;   // element offset within 32-wide K tile

  // global byte offsets (advance 64 B per 32-wide K step)
  unsigned gx = ((unsigned)(m0 + ldrow) * Ddim + ldchnk) * 2u;
  unsigned gw = ((unsigned)(n0 + ldrow) * Ddim + ldchnk) * 2u;

  // stage one 32-wide K slice into buffer b (constant), then advance offsets
  auto stage = [&](int b) {
    async_b128(lds_addr32(&sX [b][ldrow     ][ldchnk]), gx,            xb);
    async_b128(lds_addr32(&sX [b][ldrow + 64][ldchnk]), gx + 131072u,  xb);
    async_b128(lds_addr32(&sWg[b][ldrow     ][ldchnk]), gw,            wgb);
    async_b128(lds_addr32(&sWh[b][ldrow     ][ldchnk]), gw,            whb);
    gx += 64u; gw += 64u;
  };

  // consume buffer b (constant): 6 frag loads, 8 WMMAs
  auto compute = [&](int b) {
    v16bf af[2], bgf[2], bhf[2];
#pragma unroll
    for (int mi = 0; mi < 2; ++mi)
      af[mi] = load_afrag(&sX[b][wm * 32 + mi * 16 + r][0], half);
#pragma unroll
    for (int ni = 0; ni < 2; ++ni) {
      bgf[ni] = load_bfrag(&sWg[b][wn * 32 + ni * 16 + r][0], half);
      bhf[ni] = load_bfrag(&sWh[b][wn * 32 + ni * 16 + r][0], half);
    }
#pragma unroll
    for (int mi = 0; mi < 2; ++mi)
#pragma unroll
      for (int ni = 0; ni < 2; ++ni) {
        accg[mi][ni] = __builtin_amdgcn_wmma_f32_16x16x32_bf16(
            false, af[mi], false, bgf[ni], (short)0, accg[mi][ni], false, false);
        acch[mi][ni] = __builtin_amdgcn_wmma_f32_16x16x32_bf16(
            false, af[mi], false, bhf[ni], (short)0, acch[mi][ni], false, false);
      }
  };

  // 32 K-steps total: prologue + 15x2 pipelined + 2-step tail
  stage(0);
  for (int it = 0; it < 15; ++it) {
    stage(1);
    asm volatile("s_wait_asynccnt 0x4" ::: "memory");
    __syncthreads();
    compute(0);
    __syncthreads();
    stage(0);
    asm volatile("s_wait_asynccnt 0x4" ::: "memory");
    __syncthreads();
    compute(1);
    __syncthreads();
  }
  stage(1);
  asm volatile("s_wait_asynccnt 0x4" ::: "memory");
  __syncthreads();
  compute(0);
  __syncthreads();
  asm volatile("s_wait_asynccnt 0x0" ::: "memory");
  __syncthreads();
  compute(1);

  // Epilogue: C/D layout -> lane r = column, M = half*8 + vgpr j
#pragma unroll
  for (int ni = 0; ni < 2; ++ni) {
    const int n   = n0 + wn * 32 + ni * 16 + r;
    const float bgv = bg[n];
    const float bhv = bh[n];
#pragma unroll
    for (int mi = 0; mi < 2; ++mi) {
      const int mbase = m0 + wm * 32 + mi * 16 + half * 8;
#pragma unroll
      for (int j = 0; j < 8; ++j) {
        float tg = accg[mi][ni][j] + bgv;
        float tc = acch[mi][ni][j] + bhv;
        float g  = 1.0f / (1.0f + __expf(-tg));
        float e2 = __expf(2.0f * tc);
        float c  = 1.0f - 2.0f / (e2 + 1.0f);       // tanh
        size_t idx = (size_t)(mbase + j) * Ddim + n;
        aG[idx]   = f2bf(g);
        bOut[idx] = (1.0f - g) * c;
      }
    }
  }
}

// ---------------- Parallel scan over L (3 passes, 64 chunks of 64) ----------------
// Pass 1: per-chunk affine composition (A = prod a, B = chunk scan from identity)
__global__ __launch_bounds__(256) void scan_pass1(const unsigned short* __restrict__ aG,
                                                  const float* __restrict__ bOut,
                                                  float* __restrict__ sumA,
                                                  float* __restrict__ sumB) {
  int t = blockIdx.x * 256 + threadIdx.x;          // 0 .. B*NCHUNK*D-1
  int d = t & (Ddim - 1);
  int u = t >> 10;                                  // b*NCHUNK + ch
  size_t idx = (size_t)u * CLEN * Ddim + d;         // b*L + ch*CLEN row start
  float A = 1.0f, Bc = 0.0f;
  for (int i = 0; i < CLEN; ++i) {
    if (i + 8 < CLEN) __builtin_prefetch(aG + idx + (size_t)8 * Ddim, 0, 0);
    float a  = bf2f(aG[idx]);
    float bb = bOut[idx];
    A  = a * A;
    Bc = a * Bc + bb;
    idx += Ddim;
  }
  sumA[t] = A;
  sumB[t] = Bc;
}

// Pass 2: sequential combine across chunks, seeding each chunk's starting h
__global__ __launch_bounds__(256) void scan_pass2(const float* __restrict__ sumA,
                                                  const float* __restrict__ sumB,
                                                  const float* __restrict__ hidden,
                                                  float* __restrict__ hstart) {
  int t = blockIdx.x * 256 + threadIdx.x;          // 0 .. B*D-1
  int d = t & (Ddim - 1);
  int b = t >> 10;
  float h = hidden[t];
  for (int ch = 0; ch < NCHUNK; ++ch) {
    int s = (b * NCHUNK + ch) * Ddim + d;
    hstart[s] = h;
    h = sumA[s] * h + sumB[s];
  }
}

// Pass 3: replay chunk with real starting h, write outputs (in-place over bOut)
__global__ __launch_bounds__(256) void scan_pass3(const unsigned short* __restrict__ aG,
                                                  float* __restrict__ bOut,
                                                  const float* __restrict__ hstart) {
  int t = blockIdx.x * 256 + threadIdx.x;
  int d = t & (Ddim - 1);
  int u = t >> 10;
  size_t idx = (size_t)u * CLEN * Ddim + d;
  float h = hstart[t];
  for (int i = 0; i < CLEN; ++i) {
    if (i + 8 < CLEN) __builtin_prefetch(aG + idx + (size_t)8 * Ddim, 0, 0);
    float a  = bf2f(aG[idx]);
    float bb = bOut[idx];
    h = a * h + bb;
    bOut[idx] = h;
    idx += Ddim;
  }
}

extern "C" void kernel_launch(void* const* d_in, const int* in_sizes, int n_in,
                              void* d_out, int out_size, void* d_ws, size_t ws_size,
                              hipStream_t stream) {
  const float* x      = (const float*)d_in[0];   // (8,4096,1024)
  const float* hidden = (const float*)d_in[1];   // (8,1024)
  const float* Wg     = (const float*)d_in[2];   // (1024,1024)
  const float* bg     = (const float*)d_in[3];
  const float* Wh     = (const float*)d_in[4];
  const float* bh     = (const float*)d_in[5];
  float* out = (float*)d_out;

  // workspace layout (bytes): total ~138 MB
  char* ws = (char*)d_ws;
  unsigned short* xb     = (unsigned short*)(ws);                     // 64 MB
  unsigned short* wgb    = (unsigned short*)(ws + 67108864);          //  2 MB
  unsigned short* whb    = (unsigned short*)(ws + 69206016);          //  2 MB
  unsigned short* aG     = (unsigned short*)(ws + 71303168);          // 64 MB
  float*          sumA   = (float*)(ws + 138412032);                  //  2 MB
  float*          sumB   = (float*)(ws + 140509184);                  //  2 MB
  float*          hstart = (float*)(ws + 142606336);                  //  2 MB

  const int NX = Bdim * Ldim * Ddim;        // 33,554,432
  const int NW = Ddim * Ddim;               //  1,048,576

  cvt4_kernel<<<(NX / 4 + 255) / 256, 256, 0, stream>>>(x,  xb,  NX / 4);
  cvt4_kernel<<<(NW / 4 + 255) / 256, 256, 0, stream>>>(Wg, wgb, NW / 4);
  cvt4_kernel<<<(NW / 4 + 255) / 256, 256, 0, stream>>>(Wh, whb, NW / 4);

  dim3 ggrid(Ddim / 64, (Bdim * Ldim) / 128);   // (16, 256)
  gemm_gate_cand<<<ggrid, 256, 0, stream>>>(xb, wgb, whb, bg, bh, aG, out);

  const int NS = Bdim * NCHUNK * Ddim;          // 524,288
  scan_pass1<<<NS / 256, 256, 0, stream>>>(aG, out, sumA, sumB);
  scan_pass2<<<(Bdim * Ddim) / 256, 256, 0, stream>>>(sumA, sumB, hidden, hstart);
  scan_pass3<<<NS / 256, 256, 0, stream>>>(aG, out, hstart);
}
